// sde_63625645523680
// MI455X (gfx1250) — matrix operand order repaired and checked by reading, not compiled, not run on hardware
//
#include <hip/hip_runtime.h>
#include <stdint.h>

// ---------------------------------------------------------------------------
// SDE scan: 512 sequential steps, batch-2048 independent recurrences.
// Latency-optimal tiling: BTILE=16 (one WMMA M-tile) -> 128 workgroups.
// Per workgroup: persistent time loop, two 3-layer MLPs per step computed
// with v_wmma_f32_16x16x32_f16. Weight B-fragments live in VGPRs for the
// whole kernel. Activations pass through LDS between layers. z_t is staged
// with the CDNA5 async global->LDS path (ASYNCcnt), overlapped with layer-1
// compute. 4 barriers per step (combine+shift fused, one thread per row).
// ---------------------------------------------------------------------------

#define DIMN     10
#define PASTLEN  3
#define HID      128
#define TSTEPS   512
#define BATCH    2048
#define BTILE    16       // batch rows per workgroup == one WMMA M-tile
#define NWAVE    8
#define NTHREADS 256
#define DIFFW    112      // 100 diff cols padded to 7 tiles of 16

typedef __attribute__((ext_vector_type(16))) _Float16 v16h;
typedef __attribute__((ext_vector_type(8)))  float    v8f;

union FragH {
    v16h     v;
    _Float16 h[16];
    uint32_t u[8];
};

__device__ __forceinline__ v8f wmma_f16(v16h a, v16h b, v8f c) {
    // D = A(16x32,f16) x B(32x16,f16) + C(16x16,f32)
    return __builtin_amdgcn_wmma_f32_16x16x32_f16(
        /*neg_a=*/false, a, /*neg_b=*/false, b,
        /*c_mod=*/(short)0, c, /*reuse_a=*/false, /*reuse_b=*/false);
}

__device__ __forceinline__ void wait_asynccnt0() {
#if __has_builtin(__builtin_amdgcn_s_wait_asynccnt)
    __builtin_amdgcn_s_wait_asynccnt(0);
#else
    asm volatile("s_wait_asynccnt 0x0" ::: "memory");
#endif
}

// A-matrix 16x32 f16 lane layout (ISA 7.12.2):
//   lanes 0-15: M=lane,   K = {0..7, 16..23} ; lanes 16-31: M=lane-16, K = {8..15, 24..31}
//   VGPR r holds the contiguous K-pair at Kbase(r,g) = 16*(r/4) + 8*g + 2*(r%4)
__device__ __forceinline__ v16h load_a_frag(const _Float16* buf, int rowStride,
                                            int colBase, int lane) {
    FragH f;
    const int g   = lane >> 4;
    const int row = lane & 15;
    #pragma unroll
    for (int r = 0; r < 8; ++r) {
        const int kb = ((r >> 2) << 4) + (g << 3) + ((r & 3) << 1);
        f.u[r] = *reinterpret_cast<const uint32_t*>(buf + row * rowStride + colBase + kb);
    }
    return f.v;
}

// B-matrix 32x16 f16 lane layout (matches A): lanes 0-15: N=lane, K = {0..7, 16..23} ; lanes 16-31: N=lane-16, K = {8..15, 24..31}.
// Built straight from the global f32 weight W[k*ldN + n], zero-padded outside (Krows,Ncols).
__device__ __forceinline__ v16h load_b_frag_global(const float* W, int ldN,
                                                   int Krows, int Ncols,
                                                   int kBase, int nBase, int lane) {
    FragH f;
    const int g = lane >> 4;
    const int n = nBase + (lane & 15);
    #pragma unroll
    for (int v = 0; v < 8; ++v) {
        const int k0 = kBase + ((v >> 2) << 4) + (g << 3) + ((v & 3) << 1);
        float a = 0.f, b = 0.f;
        if (n < Ncols) {
            if (k0     < Krows) a = W[(size_t)k0       * ldN + n];
            if (k0 + 1 < Krows) b = W[(size_t)(k0 + 1) * ldN + n];
        }
        f.h[2 * v]     = (_Float16)a;
        f.h[2 * v + 1] = (_Float16)b;
    }
    return f.v;
}

// C-matrix f32 16x16: VGPR r, lane L -> (M = r + 8*(L/16), N = L%16)
__device__ __forceinline__ void store_h_relu(_Float16* h, int ldn,
                                             int colBase, v8f c, float bias, int lane) {
    const int g = lane >> 4, n = lane & 15;
    #pragma unroll
    for (int r = 0; r < 8; ++r) {
        float v = fmaxf(c[r] + bias, 0.f);
        h[(r + 8 * g) * ldn + colBase + n] = (_Float16)v;
    }
}

__global__ __launch_bounds__(NTHREADS)
void sde_scan_kernel(const float* __restrict__ init_state, const float* __restrict__ z,
                     const float* __restrict__ dw1, const float* __restrict__ db1,
                     const float* __restrict__ dw2, const float* __restrict__ db2,
                     const float* __restrict__ dw3, const float* __restrict__ db3,
                     const float* __restrict__ fw1, const float* __restrict__ fb1,
                     const float* __restrict__ fw2, const float* __restrict__ fb2,
                     const float* __restrict__ fw3, const float* __restrict__ fb3,
                     float* __restrict__ out) {
    __shared__ _Float16 xbuf[BTILE * 32];        // state window, cols 0..29 valid, 30..31 = 0
    __shared__ _Float16 h1d[BTILE * HID];
    __shared__ _Float16 h1f[BTILE * HID];
    __shared__ _Float16 h2d[BTILE * HID];
    __shared__ _Float16 h2f[BTILE * HID];
    __shared__ float    diffbuf[BTILE * DIFFW];  // f32 for the einsum
    __shared__ float    driftbuf[BTILE * 16];
    __shared__ float    lastbuf[BTILE * 16];     // f32 carry of the additive chain
    __shared__ float    zbuf[BTILE * DIMN];

    const int tid    = threadIdx.x;
    const int lane   = tid & 31;
    const int wave   = tid >> 5;
    const int wgBase = blockIdx.x * BTILE;

    // ---- persistent per-wave weight fragments (registers, loaded once) ----
    // L1/L2: this wave owns combined n-tile `wave` of drift (j=0) and of diff (j=1).
    // L3:    wave 0 -> drift output tile; waves 1..7 -> diff column tiles 0..6.
    v16h w1B[2], w2B[2][4], w3B[4];
    float b1v[2], b2v[2], b3v;
    {
        const int n = lane & 15;
        w1B[0] = load_b_frag_global(dw1, HID, PASTLEN * DIMN, HID, 0, wave * 16, lane);
        w1B[1] = load_b_frag_global(fw1, HID, PASTLEN * DIMN, HID, 0, wave * 16, lane);
        #pragma unroll
        for (int kt = 0; kt < 4; ++kt) {
            w2B[0][kt] = load_b_frag_global(dw2, HID, HID, HID, kt * 32, wave * 16, lane);
            w2B[1][kt] = load_b_frag_global(fw2, HID, HID, HID, kt * 32, wave * 16, lane);
        }
        if (wave == 0) {
            #pragma unroll
            for (int kt = 0; kt < 4; ++kt)
                w3B[kt] = load_b_frag_global(dw3, DIMN, HID, DIMN, kt * 32, 0, lane);
            b3v = (n < DIMN) ? db3[n] : 0.f;
        } else {
            #pragma unroll
            for (int kt = 0; kt < 4; ++kt)
                w3B[kt] = load_b_frag_global(fw3, DIMN * DIMN, HID, DIMN * DIMN,
                                             kt * 32, (wave - 1) * 16, lane);
            const int c = (wave - 1) * 16 + n;
            b3v = (c < DIMN * DIMN) ? fb3[c] : 0.f;
        }
        b1v[0] = db1[wave * 16 + n];
        b1v[1] = fb1[wave * 16 + n];
        b2v[0] = db2[wave * 16 + n];
        b2v[1] = fb2[wave * 16 + n];
    }

    // ---- init state window (f16) and f32 carry ----
    for (int idx = tid; idx < BTILE * 32; idx += NTHREADS) {
        const int b = idx >> 5, c = idx & 31;
        _Float16 v = (_Float16)0.f;
        if (c < PASTLEN * DIMN) {
            const int p = c / DIMN, d = c % DIMN;
            v = (_Float16)init_state[((size_t)p * BATCH + wgBase + b) * DIMN + d];
        }
        xbuf[idx] = v;
    }
    for (int idx = tid; idx < BTILE * 16; idx += NTHREADS) {
        const int b = idx >> 4, i = idx & 15;
        lastbuf[idx] = (i < DIMN)
            ? init_state[((size_t)2 * BATCH + wgBase + b) * DIMN + i] : 0.f;
    }
    __syncthreads();

    for (int t = 0; t < TSTEPS; ++t) {
        // ---- z_t: async global->LDS copy (no VGPR round-trip, ASYNCcnt).
        // Waves 0..4 exactly cover the 160 elements (full-EXEC waves).
        // Latency overlaps all of layer-1; retired before the first barrier.
        if (tid < BTILE * DIMN) {
            const uint32_t ldsAddr = (uint32_t)(uintptr_t)(zbuf + tid);
            const uint64_t gAddr   =
                (uint64_t)(uintptr_t)(z + ((size_t)t * BATCH + wgBase) * DIMN + tid);
            asm volatile("global_load_async_to_lds_b32 %0, %1, off"
                         :: "v"(ldsAddr), "v"(gAddr) : "memory");
        }
        if (t + 1 < TSTEPS)   // warm L2 for the next step's async copy
            __builtin_prefetch(&z[((size_t)(t + 1) * BATCH + wgBase) * DIMN + (tid & 63)], 0, 1);

        // ---- layer 1: h1 = relu(x @ W1 + b1), K=32 (30 padded) ----
        {
            v16h a = load_a_frag(xbuf, 32, 0, lane);
            #pragma unroll
            for (int j = 0; j < 2; ++j) {
                v8f c = {};
                c = wmma_f16(a, w1B[j], c);
                store_h_relu((j == 0) ? h1d : h1f, HID, wave * 16, c, b1v[j], lane);
            }
        }
        wait_asynccnt0();     // zbuf complete before the barrier publishes it
        __syncthreads();

        // ---- layer 2: h2 = relu(h1 @ W2 + b2), K=128 ----
        #pragma unroll
        for (int j = 0; j < 2; ++j) {
            const _Float16* hin = (j == 0) ? h1d : h1f;
            v8f c = {};
            #pragma unroll
            for (int kt = 0; kt < 4; ++kt) {
                v16h a = load_a_frag(hin, HID, kt * 32, lane);
                c = wmma_f16(a, w2B[j][kt], c);
            }
            store_h_relu((j == 0) ? h2d : h2f, HID, wave * 16, c, b2v[j], lane);
        }
        __syncthreads();

        // ---- layer 3: drift (wave 0) / diff (waves 1..7), f32 out, no relu ----
        {
            const _Float16* hin = (wave == 0) ? h2d : h2f;
            v8f c = {};
            #pragma unroll
            for (int kt = 0; kt < 4; ++kt) {
                v16h a = load_a_frag(hin, HID, kt * 32, lane);
                c = wmma_f16(a, w3B[kt], c);
            }
            const int g = lane >> 4, n = lane & 15;
            if (wave == 0) {
                #pragma unroll
                for (int r = 0; r < 8; ++r)
                    driftbuf[(r + 8 * g) * 16 + n] = c[r] + b3v;
            } else {
                #pragma unroll
                for (int r = 0; r < 8; ++r)
                    diffbuf[(r + 8 * g) * DIFFW + (wave - 1) * 16 + n] = c[r] + b3v;
            }
        }
        __syncthreads();

        // ---- combine + shift, fused: one thread owns one batch row ----
        // nxt_i = last_i + drift_i + sum_j diff[i,j] * z_j ; then shift the
        // 3-deep state window for this row (single owner -> race-free).
        if (tid < BTILE) {
            const int b = tid;
            float zv[DIMN];
            #pragma unroll
            for (int j = 0; j < DIMN; ++j) zv[j] = zbuf[b * DIMN + j];

            float nxt[DIMN];
            #pragma unroll
            for (int i = 0; i < DIMN; ++i) {
                float s = lastbuf[b * 16 + i] + driftbuf[b * 16 + i];
                #pragma unroll
                for (int j = 0; j < DIMN; ++j)
                    s += diffbuf[b * DIFFW + i * DIMN + j] * zv[j];
                nxt[i] = s;
            }

            float* orow = out + ((size_t)t * BATCH + wgBase + b) * DIMN;
            #pragma unroll
            for (int i = 0; i < DIMN; ++i) orow[i] = nxt[i];

            // shift window: cols[0..19] <- cols[10..29], cols[20..29] <- nxt
            uint32_t tmp[10];
            const uint32_t* src = reinterpret_cast<const uint32_t*>(xbuf + b * 32 + 10);
            #pragma unroll
            for (int r = 0; r < 10; ++r) tmp[r] = src[r];
            uint32_t* dst = reinterpret_cast<uint32_t*>(xbuf + b * 32);
            #pragma unroll
            for (int r = 0; r < 10; ++r) dst[r] = tmp[r];
            #pragma unroll
            for (int i = 0; i < DIMN; ++i) {
                lastbuf[b * 16 + i] = nxt[i];
                xbuf[b * 32 + 20 + i] = (_Float16)nxt[i];
            }
        }
        __syncthreads();
    }
}

extern "C" void kernel_launch(void* const* d_in, const int* in_sizes, int n_in,
                              void* d_out, int out_size, void* d_ws, size_t ws_size,
                              hipStream_t stream) {
    (void)in_sizes; (void)n_in; (void)out_size; (void)d_ws; (void)ws_size;
    const float* init_state = (const float*)d_in[0];
    const float* z          = (const float*)d_in[1];
    const float* dw1 = (const float*)d_in[2];
    const float* db1 = (const float*)d_in[3];
    const float* dw2 = (const float*)d_in[4];
    const float* db2 = (const float*)d_in[5];
    const float* dw3 = (const float*)d_in[6];
    const float* db3 = (const float*)d_in[7];
    const float* fw1 = (const float*)d_in[8];
    const float* fb1 = (const float*)d_in[9];
    const float* fw2 = (const float*)d_in[10];
    const float* fb2 = (const float*)d_in[11];
    const float* fw3 = (const float*)d_in[12];
    const float* fb3 = (const float*)d_in[13];
    float* out = (float*)d_out;

    dim3 grid(BATCH / BTILE);   // 128 workgroups, each owning 16 batch rows
    sde_scan_kernel<<<grid, NTHREADS, 0, stream>>>(
        init_state, z, dw1, db1, dw2, db2, dw3, db3,
        fw1, fb1, fw2, fb2, fw3, fb3, out);
}
